// CAGroup3DRoIHead_23922967838993
// MI455X (gfx1250) — matrix operand order repaired and checked by reading, not compiled, not run on hardware
//
#include <hip/hip_runtime.h>
#include <math.h>

// ---------------- problem constants (match reference) ----------------
#define GEXT   384
#define HALFG  192
#define SZC    384
#define SYZC   (384*384)
#define SXYZC  (384*384*384)
#define NOFF   125          // 5^3 conv offsets
#define PGRID  343          // 7^3 cells per ROI
#define CH     128          // channels
#define TS_LOG 19
#define TSZ    (1u << TS_LOG)
#define EMPTY_ENTRY (~0ull)

typedef __attribute__((ext_vector_type(16))) _Float16 v16h;
typedef __attribute__((ext_vector_type(8)))  float    v8f;

struct __attribute__((aligned(16))) U128 { unsigned int x, y, z, w; };
union AFrag { U128 u[2]; v16h h; };

// ---------------- CDNA5 async memory->LDS helpers ----------------
// GVS mode: lds[vdst + i] = mem[saddr + vaddr + i]; tracked by ASYNCcnt.
__device__ __forceinline__ void g2lds_b128(unsigned ldsAddr, unsigned voff, const void* sbase) {
    asm volatile("global_load_async_to_lds_b128 %0, %1, %2"
                 :: "v"(ldsAddr), "v"(voff), "s"(sbase)
                 : "memory");
}
__device__ __forceinline__ void wait_async_le8() { asm volatile("s_wait_asynccnt 0x8" ::: "memory"); }
__device__ __forceinline__ void wait_async_0()   { asm volatile("s_wait_asynccnt 0x0" ::: "memory"); }
__device__ __forceinline__ unsigned lds_lo32(const void* p) {
    return (unsigned)(unsigned long long)p;    // low 32 bits of generic addr == LDS byte address
}

// ---------------- init: hash table + reduction buffers ----------------
__global__ void k_init(unsigned long long* __restrict__ tab, float* __restrict__ red) {
    int i = blockIdx.x * blockDim.x + threadIdx.x;
    if (i < (int)TSZ) tab[i] = EMPTY_ENTRY;
    if (i < 2 * CH)   red[i] = 0.0f;           // sum[128], sumsq[128]
}

// ---------------- build hash of sparse-tensor voxel codes ----------------
__global__ void k_build(const int* __restrict__ sp_coords,
                        unsigned long long* __restrict__ tab, int Nv) {
    int i = blockIdx.x * blockDim.x + threadIdx.x;
    if (i >= Nv) return;
    int b = sp_coords[4 * i + 0];
    int x = sp_coords[4 * i + 1] / 2 + HALFG;   // coords are even multiples of CK=2
    int y = sp_coords[4 * i + 2] / 2 + HALFG;
    int z = sp_coords[4 * i + 3] / 2 + HALFG;
    int code = b * SXYZC + x * SYZC + y * SZC + z;
    unsigned long long pk = ((unsigned long long)(unsigned)code << 32) | (unsigned)i;
    unsigned h = ((unsigned)code * 2654435761u) >> (32 - TS_LOG);
    for (int t = 0; t < (int)TSZ; ++t) {
        unsigned long long old = atomicCAS(&tab[h], EMPTY_ENTRY, pk);
        if (old == EMPTY_ENTRY) return;
        h = (h + 1) & (TSZ - 1u);
    }
}

__device__ __forceinline__ int probe_tab(const unsigned long long* __restrict__ tab, int code) {
    unsigned h = ((unsigned)code * 2654435761u) >> (32 - TS_LOG);
    #pragma unroll 1
    for (int t = 0; t < 1024; ++t) {
        unsigned long long e = tab[h];
        if (e == EMPTY_ENTRY) return -1;
        if ((unsigned)(e >> 32) == (unsigned)code) return (int)(unsigned)e;
        h = (h + 1) & (TSZ - 1u);
    }
    return -1;
}

// -------- convert fp32 weights [k][cin][cout] -> f16 transposed [k][cout][cin] --------
__global__ void k_cvt(const float* __restrict__ src, _Float16* __restrict__ dst, int total) {
    int i = blockIdx.x * blockDim.x + threadIdx.x;
    if (i >= total) return;
    int k = i >> 14, r = i & 16383, d = r >> 7, c = r & 127;
    dst[((size_t)k << 14) + (d << 7) + c] = (_Float16)src[((size_t)k << 14) + (c << 7) + d];
}

// -------- plain fp32 -> f16 feature conversion --------
__global__ void k_cvtf(const float* __restrict__ src, _Float16* __restrict__ dst, int total) {
    int i = blockIdx.x * blockDim.x + threadIdx.x;
    if (i < total) dst[i] = (_Float16)src[i];
}

// ---------------- fragment helpers (16-bit A/B VGPR striping, wave32) ----------------
__device__ __forceinline__ v16h load_afrag(const _Float16* __restrict__ rowBase, int kc, int hi) {
    AFrag f;
    const _Float16* p = rowBase + kc * 32 + hi * 8;
    f.u[0] = *(const U128*)(p);
    f.u[1] = *(const U128*)(p + 16);
    return f.h;
}
__device__ __forceinline__ v16h load_bfrag(const _Float16* __restrict__ sB, int nt, int kc, int r, int hi) {
    AFrag f;
    const _Float16* p = sB + (size_t)(nt * 16 + r) * CH + kc * 32 + hi * 8;
    f.u[0] = *(const U128*)(p);
    f.u[1] = *(const U128*)(p + 16);
    return f.h;
}

// ---------------- sparse conv: async gather + async W-tiles + WMMA ----------------
__global__ __launch_bounds__(256) void k_conv(const float* __restrict__ gp,
                                              const _Float16* __restrict__ featsH,
                                              const unsigned long long* __restrict__ tab,
                                              const _Float16* __restrict__ W1h,
                                              float* __restrict__ accOut, int N) {
    __shared__ __align__(16) _Float16 sB[2][CH * CH];     // double-buffered W1[k] tile
    __shared__ __align__(16) _Float16 sA[8][16 * CH];     // per-wave A tiles

    const int tid  = threadIdx.x;
    const int wave = tid >> 5, lane = tid & 31;
    const int r = lane & 15, hi = lane >> 4;
    const int tileBase = (blockIdx.x * 8 + wave) * 16;
    const int row = tileBase + r;
    const bool rv = row < N;

    int qb = 0, qx = 0, qy = 0, qz = 0;
    if (rv) {
        qb = (int)gp[row * 4 + 0];
        float fx = gp[row * 4 + 1], fy = gp[row * 4 + 2], fz = gp[row * 4 + 3];
        qx = min(max((int)floorf(fx / 0.08f), -(HALFG - 1)), HALFG - 1) + HALFG;
        qy = min(max((int)floorf(fy / 0.08f), -(HALFG - 1)), HALFG - 1) + HALFG;
        qz = min(max((int)floorf(fz / 0.08f), -(HALFG - 1)), HALFG - 1) + HALFG;
    }

    const unsigned ldsW0 = lds_lo32(&sB[0][0]) + (unsigned)tid * 16u;
    const unsigned ldsW1 = lds_lo32(&sB[1][0]) + (unsigned)tid * 16u;
    _Float16* arow = &sA[wave][r * CH + hi * 64];
    const unsigned ldsA = lds_lo32(arow);

    v8f acc[8];
    #pragma unroll
    for (int nt = 0; nt < 8; ++nt) acc[nt] = (v8f)0.0f;

    // prologue: stage W1[0] into buffer 0 (8 async b128 per thread)
    #pragma unroll
    for (int j = 0; j < 8; ++j)
        g2lds_b128(ldsW0 + j * 4096u, (unsigned)tid * 16u + j * 4096u, W1h);

    for (int k = 0; k < NOFF; ++k) {
        __syncthreads();                    // B1: everyone done computing on both buffers

        // ---- gather A tile (async hit path / ds zero miss path) ----
        int sidx = -1;
        if (rv) {
            int dx = k / 25 - 2, dy = (k / 5) % 5 - 2, dz = k % 5 - 2;
            int tx = qx + dx, ty = qy + dy, tz = qz + dz;
            if ((unsigned)tx < GEXT && (unsigned)ty < GEXT && (unsigned)tz < GEXT)
                sidx = probe_tab(tab, qb * SXYZC + tx * SYZC + ty * SZC + tz);
        }
        if (sidx >= 0) {
            unsigned voff = (unsigned)sidx * 256u + (unsigned)hi * 128u;
            #pragma unroll
            for (int j = 0; j < 8; ++j)
                g2lds_b128(ldsA + j * 16u, voff + j * 16u, featsH);
        } else {
            U128 zz = {0u, 0u, 0u, 0u};
            #pragma unroll
            for (int j = 0; j < 8; ++j) ((U128*)arow)[j] = zz;
        }

        // ---- stage next W tile into alternate buffer, retire current ----
        if (k + 1 < NOFF) {
            const _Float16* wb = W1h + (size_t)(k + 1) * CH * CH;
            unsigned ldsw = ((k + 1) & 1) ? ldsW1 : ldsW0;
            #pragma unroll
            for (int j = 0; j < 8; ++j)
                g2lds_b128(ldsw + j * 4096u, (unsigned)tid * 16u + j * 4096u, wb);
            wait_async_le8();               // in-order retire: W(k) + gather(k) done
        } else {
            wait_async_0();
        }
        __syncthreads();                    // B2: W(k) chunks from all waves visible

        // ---- 8 N-tiles x 4 K-chunks of WMMA ----
        const _Float16* sBk = sB[k & 1];
        const _Float16* rowBase = &sA[wave][r * CH];
        v16h af[4];
        #pragma unroll
        for (int kc = 0; kc < 4; ++kc) af[kc] = load_afrag(rowBase, kc, hi);
        #pragma unroll
        for (int nt = 0; nt < 8; ++nt) {
            #pragma unroll
            for (int kc = 0; kc < 4; ++kc) {
                v16h bf = load_bfrag(sBk, nt, kc, r, hi);
                acc[nt] = __builtin_amdgcn_wmma_f32_16x16x32_f16(
                    false, af[kc], false, bf, (short)0, acc[nt], false, false);
            }
        }
    }

    // write accumulator (C/D layout: vgpr j -> M = j + hi*8, N = lane&15)
    #pragma unroll
    for (int nt = 0; nt < 8; ++nt)
        #pragma unroll
        for (int j = 0; j < 8; ++j) {
            int orow = tileBase + j + hi * 8;
            if (orow < N) accOut[(size_t)orow * CH + nt * 16 + r] = acc[nt][j];
        }
}

// ---------------- BN1 reduction over all rows ----------------
__global__ void k_reduce1(const float* __restrict__ acc, float* __restrict__ red, int N) {
    int c = threadIdx.x;                         // 128 threads = channels
    int chunk = (N + gridDim.x - 1) / gridDim.x;
    int r0 = blockIdx.x * chunk, r1 = min(N, r0 + chunk);
    float s = 0.f, q = 0.f;
    for (int rr = r0; rr < r1; ++rr) { float v = acc[(size_t)rr * CH + c]; s += v; q += v * v; }
    atomicAdd(&red[c], s);
    atomicAdd(&red[CH + c], q);
}

__global__ void k_fin1(float* __restrict__ red, int N) {
    int c = threadIdx.x;
    if (c >= CH) return;
    float n = (float)N;
    float mean = red[c] / n;
    float var  = red[CH + c] / n - mean * mean;
    red[2 * CH + c] = mean;
    red[3 * CH + c] = rsqrtf(var + 1e-5f);
}

// ---------------- pooling conv: fused BN1+ELU gather, async W2, WMMA ----------------
__global__ __launch_bounds__(256) void k_pool(const float* __restrict__ acc,
                                              const _Float16* __restrict__ W2h,
                                              const float* __restrict__ red,
                                              const float* __restrict__ g1,
                                              const float* __restrict__ b1,
                                              float* __restrict__ pooled, int R) {
    __shared__ __align__(16) _Float16 sB[2][CH * CH];
    __shared__ __align__(16) _Float16 sA[8][16 * CH];
    __shared__ float sScale[CH], sBias[CH];

    const int tid  = threadIdx.x;
    const int wave = tid >> 5, lane = tid & 31;
    const int r = lane & 15, hi = lane >> 4;
    const int roiBase = (blockIdx.x * 8 + wave) * 16;
    const int roi = roiBase + r;
    const bool rv = roi < R;

    if (tid < CH) {
        float sc = red[3 * CH + tid] * g1[tid];
        sScale[tid] = sc;
        sBias[tid]  = b1[tid] - red[2 * CH + tid] * sc;
    }

    const unsigned ldsW0 = lds_lo32(&sB[0][0]) + (unsigned)tid * 16u;
    const unsigned ldsW1 = lds_lo32(&sB[1][0]) + (unsigned)tid * 16u;

    v8f acc8[8];
    #pragma unroll
    for (int nt = 0; nt < 8; ++nt) acc8[nt] = (v8f)0.0f;

    // prologue: stage W2[0]
    #pragma unroll
    for (int j = 0; j < 8; ++j)
        g2lds_b128(ldsW0 + j * 4096u, (unsigned)tid * 16u + j * 4096u, W2h);

    for (int p = 0; p < PGRID; ++p) {
        __syncthreads();                    // B1

        if (p + 1 < PGRID) {
            const _Float16* wb = W2h + (size_t)(p + 1) * CH * CH;
            unsigned ldsw = ((p + 1) & 1) ? ldsW1 : ldsW0;
            #pragma unroll
            for (int j = 0; j < 8; ++j)
                g2lds_b128(ldsw + j * 4096u, (unsigned)tid * 16u + j * 4096u, wb);
            wait_async_le8();               // W(p) retired
        } else {
            wait_async_0();
        }
        __syncthreads();                    // B2

        // gather with fused BN1 + ELU (needs VALU transform)
        _Float16* arow = &sA[wave][r * CH + hi * 64];
        if (rv) {
            const float4* src = (const float4*)(acc + ((size_t)roi * PGRID + p) * CH + hi * 64);
            #pragma unroll
            for (int j = 0; j < 16; ++j) {
                float4 v = src[j];
                #pragma unroll
                for (int cc = 0; cc < 4; ++cc) {
                    int c = hi * 64 + j * 4 + cc;
                    float xv = (cc == 0 ? v.x : cc == 1 ? v.y : cc == 2 ? v.z : v.w);
                    xv = xv * sScale[c] + sBias[c];
                    xv = xv > 0.f ? xv : expm1f(xv);      // ELU
                    arow[j * 4 + cc] = (_Float16)xv;
                }
            }
        } else {
            U128 zz = {0u, 0u, 0u, 0u};
            #pragma unroll
            for (int j = 0; j < 8; ++j) ((U128*)arow)[j] = zz;
        }

        const _Float16* sBp = sB[p & 1];
        const _Float16* rowBase = &sA[wave][r * CH];
        v16h af[4];
        #pragma unroll
        for (int kc = 0; kc < 4; ++kc) af[kc] = load_afrag(rowBase, kc, hi);
        #pragma unroll
        for (int nt = 0; nt < 8; ++nt) {
            #pragma unroll
            for (int kc = 0; kc < 4; ++kc) {
                v16h bf = load_bfrag(sBp, nt, kc, r, hi);
                acc8[nt] = __builtin_amdgcn_wmma_f32_16x16x32_f16(
                    false, af[kc], false, bf, (short)0, acc8[nt], false, false);
            }
        }
    }

    #pragma unroll
    for (int nt = 0; nt < 8; ++nt)
        #pragma unroll
        for (int j = 0; j < 8; ++j) {
            int orow = roiBase + j + hi * 8;
            if (orow < R) pooled[(size_t)orow * CH + nt * 16 + r] = acc8[nt][j];
        }
}

// ---------------- final BN over pooled rows ----------------
__global__ void k_bn2(const float* __restrict__ pooled, const float* __restrict__ gamma2,
                      const float* __restrict__ beta2, float* __restrict__ out, int R) {
    int c = blockIdx.x;            // one block per channel
    int t = threadIdx.x;           // 256 threads
    __shared__ float ss[256], sq[256];
    __shared__ float sm, si;
    float s = 0.f, q = 0.f;
    for (int b = t; b < R; b += 256) { float v = pooled[(size_t)b * CH + c]; s += v; q += v * v; }
    ss[t] = s; sq[t] = q;
    __syncthreads();
    for (int w = 128; w > 0; w >>= 1) {
        if (t < w) { ss[t] += ss[t + w]; sq[t] += sq[t + w]; }
        __syncthreads();
    }
    if (t == 0) {
        float mean = ss[0] / (float)R;
        float var  = sq[0] / (float)R - mean * mean;
        sm = mean; si = rsqrtf(var + 1e-5f);
    }
    __syncthreads();
    float g = gamma2[c], be = beta2[c];
    for (int b = t; b < R; b += 256)
        out[(size_t)b * CH + c] = (pooled[(size_t)b * CH + c] - sm) * si * g + be;
}

// ---------------- host launcher ----------------
extern "C" void kernel_launch(void* const* d_in, const int* in_sizes, int n_in,
                              void* d_out, int out_size, void* d_ws, size_t ws_size,
                              hipStream_t stream) {
    (void)n_in; (void)out_size; (void)ws_size;
    const int*   sp_coords   = (const int*)  d_in[0];
    const float* sp_feats    = (const float*)d_in[1];
    const float* grid_points = (const float*)d_in[2];
    const float* W1          = (const float*)d_in[3];
    const float* gamma1      = (const float*)d_in[4];
    const float* beta1       = (const float*)d_in[5];
    const float* W2          = (const float*)d_in[6];
    const float* gamma2      = (const float*)d_in[7];
    const float* beta2       = (const float*)d_in[8];

    const int Nv = in_sizes[0] / 4;
    const int N  = in_sizes[2] / 4;
    const int K1 = in_sizes[3] / (CH * CH);   // 125
    const int K2 = in_sizes[6] / (CH * CH);   // 343
    const int R  = N / PGRID;                 // 512 ROIs
    float* out = (float*)d_out;

    // workspace layout
    char* p = (char*)d_ws;
    auto take = [&](size_t bytes) { char* r = p; p += (bytes + 255) & ~(size_t)255; return r; };
    unsigned long long* tab    = (unsigned long long*)take((size_t)TSZ * 8);
    _Float16*           W1h    = (_Float16*)take((size_t)K1 * CH * CH * 2);
    _Float16*           W2h    = (_Float16*)take((size_t)K2 * CH * CH * 2);
    _Float16*           featsH = (_Float16*)take((size_t)Nv * CH * 2);
    float*              accW   = (float*)take((size_t)N * CH * 4);
    float*              pool   = (float*)take((size_t)R * CH * 4);
    float*              red    = (float*)take((size_t)4 * CH * 4);

    k_init  <<<(TSZ + 255) / 256, 256, 0, stream>>>(tab, red);
    k_build <<<(Nv + 255) / 256, 256, 0, stream>>>(sp_coords, tab, Nv);
    k_cvt   <<<(K1 * CH * CH + 255) / 256, 256, 0, stream>>>(W1, W1h, K1 * CH * CH);
    k_cvt   <<<(K2 * CH * CH + 255) / 256, 256, 0, stream>>>(W2, W2h, K2 * CH * CH);
    k_cvtf  <<<(Nv * CH + 255) / 256, 256, 0, stream>>>(sp_feats, featsH, Nv * CH);
    k_conv  <<<(N + 127) / 128, 256, 0, stream>>>(grid_points, featsH, tab, W1h, accW, N);
    k_reduce1<<<512, 128, 0, stream>>>(accW, red, N);
    k_fin1  <<<1, 128, 0, stream>>>(red, N);
    k_pool  <<<(R + 127) / 128, 256, 0, stream>>>(accW, W2h, red, gamma1, beta1, pool, R);
    k_bn2   <<<CH, 256, 0, stream>>>(pool, gamma2, beta2, out, R);
}
